// GResNetBlock_43894565765752
// MI455X (gfx1250) — matrix operand order repaired and checked by reading, not compiled, not run on hardware
//
#include <hip/hip_runtime.h>

typedef __bf16 bf16;
typedef __attribute__((ext_vector_type(16))) __bf16 v16bf;
typedef __attribute__((ext_vector_type(8)))  __bf16 v8bf;
typedef __attribute__((ext_vector_type(8)))  float  v8f;
typedef unsigned int u32x4 __attribute__((ext_vector_type(4)));
typedef int          i32x4 __attribute__((ext_vector_type(4)));
typedef int          i32x8 __attribute__((ext_vector_type(8)));

#define SIGMA_C 0.5f
#define EPS_C   1e-5f

// Problem constants (fixed by setup_inputs)
#define Bsz 2
#define Ci  16
#define Cout 32
#define Gn  12
#define Rn  12
#define Sdim 32
#define S3  (Sdim*Sdim*Sdim)        // 32768
#define NG  (Cout*Gn)               // 384  output rows of group-mix GEMMs
#define K1n (Ci*Gn)                 // 192
#define K2n (Cout*Gn)               // 384
#define VOLY ((size_t)Bsz*NG*S3)    // 25,165,824 elements of y/z
#define VOLX ((size_t)Bsz*K1n*S3)   // 12,582,912 elements of x
#define SLAB ((size_t)Gn*S3)        // 393,216 per (b,co)

// padded LDS geometry: 4 pad dwords inserted after every 64 data dwords
// (keeps every 4-dword-aligned data offset 16B-aligned after padding)
#define PADOFF(d) ((d) + 4 * ((d) >> 6))
#define LDS_DW (32 * (K2n / 2) + 4 * ((32 * (K2n / 2)) / 64))   // 6528 dwords

static __device__ __forceinline__ bf16 f2bf(float f) { return (bf16)f; }

// ---------------------------------------------------------------------------
__global__ void k_zero(float* p, int n) {
    int i = blockIdx.x * blockDim.x + threadIdx.x;
    if (i < n) p[i] = 0.f;
}

// ---------------------------------------------------------------------------
// Small prep: RBF mixing weights (3 sets) + rotated 3x3x3 depthwise kernels
__global__ void k_prep_small(const float* __restrict__ H,
                             const float* __restrict__ Hr1,
                             const float* __restrict__ Hr2,
                             const float* __restrict__ Hrr,
                             const float* __restrict__ Ws1,
                             const float* __restrict__ Ws2,
                             float* __restrict__ w1, float* __restrict__ w2,
                             float* __restrict__ wr,
                             float* __restrict__ Kr1, float* __restrict__ Kr2) {
    const int t = threadIdx.x;
    for (int task = t; task < 3 * Gn * Gn; task += 256) {
        int set = task / (Gn * Gn);
        int og  = task % (Gn * Gn);
        int o = og / Gn, g = og % Gn;
        const float* Hr = (set == 0) ? Hr1 : ((set == 1) ? Hr2 : Hrr);
        float* wout     = (set == 0) ? w1  : ((set == 1) ? w2  : wr);
        float Rr[9];
        for (int i2 = 0; i2 < 3; ++i2)
            for (int j = 0; j < 3; ++j) {
                float s = 0.f;
                for (int k = 0; k < 3; ++k) s += H[o*9 + k*3 + i2] * H[g*9 + k*3 + j];
                Rr[i2*3 + j] = s;
            }
        float wv[Rn]; float tot = 0.f;
        for (int r = 0; r < Rn; ++r) {
            float tr = 0.f;
            for (int e = 0; e < 9; ++e) tr += Rr[e] * Hr[r*9 + e];
            float c = (tr - 1.f) * 0.5f;
            c = fminf(fmaxf(c, -1.f + 1e-6f), 1.f - 1e-6f);
            float d = acosf(c);
            float v = __expf(-(d * d) / (2.f * SIGMA_C * SIGMA_C));
            wv[r] = v; tot += v;
        }
        float rin = 1.f / tot;
        for (int r = 0; r < Rn; ++r) wout[og * Rn + r] = wv[r] * rin;
    }
    for (int task = t; task < 2 * Cout * Gn * 27; task += 256) {
        int which = task / (Cout * Gn * 27);
        int rem   = task % (Cout * Gn * 27);
        int c   = rem / (Gn * 27);
        int g   = (rem / 27) % Gn;
        int tap = rem % 27;
        int dz = tap / 9, dy = (tap / 3) % 3, dx = tap % 3;
        float c0 = (float)dz - 1.f, c1 = (float)dy - 1.f, c2 = (float)dx - 1.f;
        const float* Hg = H + g * 9;
        float p0 = c0*Hg[0] + c1*Hg[3] + c2*Hg[6] + 1.f;
        float p1 = c0*Hg[1] + c1*Hg[4] + c2*Hg[7] + 1.f;
        float p2 = c0*Hg[2] + c1*Hg[5] + c2*Hg[8] + 1.f;
        float f0 = floorf(p0), f1 = floorf(p1), f2 = floorf(p2);
        float t0 = p0 - f0, t1 = p1 - f1, t2 = p2 - f2;
        int i0 = (int)f0, i1 = (int)f1, i2 = (int)f2;
        const float* Wsrc = ((which == 0) ? Ws1 : Ws2) + c * 27;
        float acc = 0.f;
        for (int cz = 0; cz < 2; ++cz)
            for (int cy = 0; cy < 2; ++cy)
                for (int cx = 0; cx < 2; ++cx) {
                    int az = i0 + cz, ay = i1 + cy, ax = i2 + cx;
                    bool valid = (az >= 0 && az <= 2 && ay >= 0 && ay <= 2 &&
                                  ax >= 0 && ax <= 2);
                    int bz = min(max(az, 0), 2), by = min(max(ay, 0), 2),
                        bx = min(max(ax, 0), 2);
                    float val = Wsrc[bz*9 + by*3 + bx];
                    float wz = cz ? t0 : 1.f - t0;
                    float wy = cy ? t1 : 1.f - t1;
                    float wx = cx ? t2 : 1.f - t2;
                    acc += val * (wz * wy * wx) * (valid ? 1.f : 0.f);
                }
        ((which == 0) ? Kr1 : Kr2)[c * Gn * 27 + g * 27 + tap] = acc;
    }
}

// ---------------------------------------------------------------------------
// Pack group-mix matrices directly in the A-fragment order k_gemm uses (contiguous k, same as B):
//   Apack[((nt*(K/32)+kc)*32 + lane)*16 + j]
//   n = nt*16 + (lane&15) ; k = kc*32 + (lane>>4)*16 + j
__global__ void k_prep_A(const float* __restrict__ Wg1, const float* __restrict__ Wg2,
                         const float* __restrict__ Wres,
                         const float* __restrict__ w1, const float* __restrict__ w2,
                         const float* __restrict__ wr,
                         bf16* __restrict__ A1, bf16* __restrict__ A2,
                         bf16* __restrict__ Ar) {
    int idx = blockIdx.x * 256 + threadIdx.x;
    const int n1 = NG * K1n;    // 73728
    const int n2 = NG * K2n;    // 147456
    const float* Wg; const float* w; bf16* A; int K; int rel;
    if (idx < n1)            { Wg = Wg1;  w = w1; A = A1; K = K1n; rel = idx; }
    else if (idx < n1 + n2)  { Wg = Wg2;  w = w2; A = A2; K = K2n; rel = idx - n1; }
    else if (idx < 2*n1+n2)  { Wg = Wres; w = wr; A = Ar; K = K1n; rel = idx - n1 - n2; }
    else return;
    int j    = rel & 15;
    int lane = (rel >> 4) & 31;
    int kc   = (rel >> 9) % (K >> 5);
    int nt   = rel / (512 * (K >> 5));
    int n = nt * 16 + (lane & 15);
    int k = kc * 32 + (lane >> 4) * 16 + j;
    int a = n / Gn, o = n % Gn;
    int c = k / Gn, g = k % Gn;
    int CiLoc = K / Gn;
    float s = 0.f;
    for (int r = 0; r < Rn; ++r)
        s += Wg[(a * CiLoc + c) * Rn + r] * w[(o * Gn + g) * Rn + r];
    A[rel] = f2bf(s);
}

// ---------------------------------------------------------------------------
// Tiled transpose+convert: x[b][k][m] (fp32) -> xbT[b][m][k] (bf16)
__global__ void k_cvtT(const float* __restrict__ X, bf16* __restrict__ T) {
    __shared__ float tile[32][33];
    const int b  = blockIdx.z;
    const int m0 = blockIdx.x * 32, k0 = blockIdx.y * 32;
    #pragma unroll
    for (int i = 0; i < 4; ++i) {
        int e = threadIdx.x + i * 256;
        int kr = e >> 5, mc = e & 31;
        tile[kr][mc] = X[((size_t)b * K1n + k0 + kr) * S3 + m0 + mc];
    }
    __syncthreads();
    #pragma unroll
    for (int i = 0; i < 4; ++i) {
        int e = threadIdx.x + i * 256;
        int mr = e >> 5, kc = e & 31;
        T[((size_t)b * S3 + m0 + mr) * K1n + k0 + kc] = f2bf(tile[kc][mr]);
    }
}

// Instance-norm apply + ReLU fused with transpose: y[b][n][m] -> t1bT[b][m][n] bf16
__global__ void k_in_transpose(const float* __restrict__ Y, const float* __restrict__ st,
                               bf16* __restrict__ T) {
    __shared__ float tile[32][33];
    const int b  = blockIdx.z;
    const int m0 = blockIdx.x * 32, n0 = blockIdx.y * 32;
    const float inv = 1.0f / (float)SLAB;
    #pragma unroll
    for (int i = 0; i < 4; ++i) {
        int e = threadIdx.x + i * 256;
        int nr = e >> 5, mc = e & 31;
        int n = n0 + nr;
        int slab = b * Cout + n / Gn;
        float mu  = st[slab * 2] * inv;
        float var = st[slab * 2 + 1] * inv - mu * mu;
        float v = (Y[((size_t)b * NG + n) * S3 + m0 + mc] - mu) * rsqrtf(var + EPS_C);
        tile[nr][mc] = v > 0.f ? v : 0.f;
    }
    __syncthreads();
    #pragma unroll
    for (int i = 0; i < 4; ++i) {
        int e = threadIdx.x + i * 256;
        int mr = e >> 5, nc = e & 31;
        T[((size_t)b * S3 + m0 + mr) * K2n + n0 + nc] = f2bf(tile[nc][mr]);
    }
}

// ---------------------------------------------------------------------------
// WMMA GEMM:  Z[b][n][m] = sum_k A[n][k] * XT[b][m][k]
// block = 8 waves; block tile 128(N) x 32(M); per-wave 16x32.
// The 32-row x K bf16 B-panel is DMA'd to LDS once per block by the TDM.
__global__ void __launch_bounds__(256)
k_gemm(const bf16* __restrict__ Apack, const bf16* __restrict__ XT,
       float* __restrict__ Z, int K) {
    __shared__ unsigned int BtDW[LDS_DW];
    const int tid = threadIdx.x, wave = tid >> 5, lane = tid & 31;
    const int row = lane & 15, hi = lane >> 4;
    const int m0  = blockIdx.x * 32;
    const int n0t = blockIdx.y * 8;                 // first n-tile of block
    const bf16* Xb = XT + ((size_t)blockIdx.z * S3 + m0) * K;  // panel start
    float*      Zb = Z + (size_t)blockIdx.z * NG * S3;
    const int halfK = K >> 1;                       // dwords per row

#if __has_builtin(__builtin_amdgcn_tensor_load_to_lds)
    if (tid < 32) {  // wave 0 issues the tensor DMA (D# per ISA ch.8)
        unsigned long long ga = (unsigned long long)Xb;
        unsigned int ldsoff = (unsigned int)(unsigned long long)(void*)BtDW;
        u32x4 g0; i32x8 g1; i32x4 gz = {0, 0, 0, 0};
        g0[0] = 1u;                                          // count=1 (valid)
        g0[1] = ldsoff;                                      // lds_addr (bytes)
        g0[2] = (unsigned int)ga;                            // global_addr[31:0]
        g0[3] = (unsigned int)((ga >> 32) & 0x1FFFFFFu) | (2u << 30); // [56:32]|type=2
        g1[0] = (1 << 16) | (1 << 20) | (5 << 22) | (3 << 25); // 2B elems, pad_en,
                                                               // 64dw interval, 4dw pad
        g1[1] = (int)((unsigned)(K & 0xFFFF) << 16);         // tensor_dim0 = K (lo16)
        g1[2] = (int)(0x8000u << 16);                        // dim0 hi | tensor_dim1=32768
        g1[3] = (int)((unsigned)(K & 0xFFFF) << 16);         // dim1 hi | tile_dim0 = K
        g1[4] = 32;                                          // tile_dim1=32, tile_dim2=0
        g1[5] = K;                                           // tensor_dim0_stride = K
        g1[6] = 0; g1[7] = 0;
#if __clang_major__ >= 23
        i32x8 gz8 = {0, 0, 0, 0, 0, 0, 0, 0};
        __builtin_amdgcn_tensor_load_to_lds(g0, g1, gz, gz, gz8, 0);
#else
        __builtin_amdgcn_tensor_load_to_lds(g0, g1, gz, gz, 0);
#endif
        __builtin_amdgcn_s_wait_tensorcnt(0);
    }
#else
    // fallback: cooperative copy, same padded layout
    for (int e = tid; e < 32 * halfK; e += 256) {
        BtDW[PADOFF(e)] = ((const unsigned int*)Xb)[e];
    }
#endif
    __syncthreads();

    v8f acc0 = {}; v8f acc1 = {};
    const int nchunks = K >> 5;
    const bf16* Ap = Apack + ((((size_t)(n0t + wave)) * nchunks) * 32 + lane) * 16;
    for (int kc = 0; kc < nchunks; ++kc) {
        // A fragment: one coalesced 32B load (pre-packed fragment order)
        const v16bf* apf = (const v16bf*)__builtin_assume_aligned(
            (const void*)(Ap + (size_t)kc * 512), 32);
        v16bf af = *apf;
        __builtin_prefetch(Ap + (size_t)(kc + 1) * 512, 0, 1);
        // B fragments from padded LDS: lane column m, 16 consecutive k
        // (every 4-dword data offset stays 16B-aligned under the padding scheme)
        int kdw = (kc << 4) + hi * 8;                 // dword offset in row
        int d0 = row * halfK + kdw;
        int d1 = (row + 16) * halfK + kdw;
        const v8bf* p0a = (const v8bf*)__builtin_assume_aligned(
            (const void*)((const char*)BtDW + 4 * PADOFF(d0)), 16);
        const v8bf* p0b = (const v8bf*)__builtin_assume_aligned(
            (const void*)((const char*)BtDW + 4 * PADOFF(d0 + 4)), 16);
        const v8bf* p1a = (const v8bf*)__builtin_assume_aligned(
            (const void*)((const char*)BtDW + 4 * PADOFF(d1)), 16);
        const v8bf* p1b = (const v8bf*)__builtin_assume_aligned(
            (const void*)((const char*)BtDW + 4 * PADOFF(d1 + 4)), 16);
        v8bf b0a = *p0a, b0b = *p0b, b1a = *p1a, b1b = *p1b;
        v16bf bf0 = __builtin_shufflevector(b0a, b0b, 0,1,2,3,4,5,6,7,8,9,10,11,12,13,14,15);
        v16bf bf1 = __builtin_shufflevector(b1a, b1b, 0,1,2,3,4,5,6,7,8,9,10,11,12,13,14,15);
        acc0 = __builtin_amdgcn_wmma_f32_16x16x32_bf16(false, af, false, bf0,
                                                       (short)0, acc0, false, false);
        acc1 = __builtin_amdgcn_wmma_f32_16x16x32_bf16(false, af, false, bf1,
                                                       (short)0, acc1, false, false);
    }
    const int outn = n0t * 16 + wave * 16 + hi * 8;
    const int outm = m0 + row;
    #pragma unroll
    for (int r = 0; r < 8; ++r) {
        Zb[(size_t)(outn + r) * S3 + outm]      = acc0[r];
        Zb[(size_t)(outn + r) * S3 + outm + 16] = acc1[r];
    }
}

// ---------------------------------------------------------------------------
// Depthwise 3x3x3 conv (cross-correlation, zero pad=1), per (b,c,g) slab
__global__ void k_dwconv(const float* __restrict__ Z, const float* __restrict__ Kr,
                         float* __restrict__ Y) {
    const int slab = blockIdx.y;
    const int cg   = slab % NG;
    const int p = blockIdx.x * blockDim.x + threadIdx.x;
    const int z = p >> 10, y = (p >> 5) & 31, x = p & 31;
    const float* src = Z + (size_t)slab * S3;
    const float* kv  = Kr + cg * 27;
    float acc = 0.f;
    #pragma unroll
    for (int t = 0; t < 27; ++t) {
        int dz = t / 9 - 1, dy = (t / 3) % 3 - 1, dx = t % 3 - 1;
        int zz = z + dz, yy = y + dy, xx = x + dx;
        bool ok = ((unsigned)zz < 32u) && ((unsigned)yy < 32u) && ((unsigned)xx < 32u);
        float v = ok ? src[(zz << 10) + (yy << 5) + xx] : 0.f;
        acc += v * kv[t];
    }
    Y[(size_t)slab * S3 + p] = acc;
}

// ---------------------------------------------------------------------------
__global__ void k_instats(const float* __restrict__ Y, float* st) {
    __shared__ float s1[256], s2[256];
    const int slab = blockIdx.y;
    const float* src = Y + (size_t)slab * SLAB;
    long base = (long)blockIdx.x * 16384;
    float a = 0.f, b = 0.f;
    for (int i = threadIdx.x; i < 16384; i += 256) {
        float v = src[base + i];
        a += v; b += v * v;
    }
    s1[threadIdx.x] = a; s2[threadIdx.x] = b;
    __syncthreads();
    for (int o = 128; o > 0; o >>= 1) {
        if (threadIdx.x < o) {
            s1[threadIdx.x] += s1[threadIdx.x + o];
            s2[threadIdx.x] += s2[threadIdx.x + o];
        }
        __syncthreads();
    }
    if (threadIdx.x == 0) {
        atomicAdd(&st[slab * 2],     s1[0]);
        atomicAdd(&st[slab * 2 + 1], s2[0]);
    }
}

// Instance-norm apply + ReLU, fp32 in place (second IN)
__global__ void k_inapply_f32(float* Y, const float* __restrict__ st) {
    long i = (long)blockIdx.x * blockDim.x + threadIdx.x;
    long stride = (long)gridDim.x * blockDim.x;
    const float inv = 1.0f / (float)SLAB;
    for (; i < (long)VOLY; i += stride) {
        int slab = (int)(i / (long)SLAB);
        float mu  = st[slab * 2] * inv;
        float var = st[slab * 2 + 1] * inv - mu * mu;
        float v = (Y[i] - mu) * rsqrtf(var + EPS_C);
        Y[i] = v > 0.f ? v : 0.f;
    }
}

// ---------------------------------------------------------------------------
__global__ void k_add_bnstats(float* Y, const float* __restrict__ R, float* bst) {
    __shared__ float s1[256], s2[256];
    const int slab = blockIdx.y;
    const int co   = slab % Cout;
    size_t base = (size_t)slab * SLAB + (size_t)blockIdx.x * 4096;
    float a = 0.f, b = 0.f;
    for (int i = threadIdx.x; i < 4096; i += 256) {
        float v = Y[base + i] + R[base + i];
        Y[base + i] = v;
        a += v; b += v * v;
    }
    s1[threadIdx.x] = a; s2[threadIdx.x] = b;
    __syncthreads();
    for (int o = 128; o > 0; o >>= 1) {
        if (threadIdx.x < o) {
            s1[threadIdx.x] += s1[threadIdx.x + o];
            s2[threadIdx.x] += s2[threadIdx.x + o];
        }
        __syncthreads();
    }
    if (threadIdx.x == 0) {
        atomicAdd(&bst[co * 2],     s1[0]);
        atomicAdd(&bst[co * 2 + 1], s2[0]);
    }
}

__global__ void k_bnapply(float* Y, const float* __restrict__ bst,
                          const float* __restrict__ gamma,
                          const float* __restrict__ beta) {
    long i = (long)blockIdx.x * blockDim.x + threadIdx.x;
    long stride = (long)gridDim.x * blockDim.x;
    const float inv = 1.0f / (float)(Bsz * SLAB);
    for (; i < (long)VOLY; i += stride) {
        int co = (int)((i / (long)SLAB) % Cout);
        float mu  = bst[co * 2] * inv;
        float var = bst[co * 2 + 1] * inv - mu * mu;
        Y[i] = (Y[i] - mu) * rsqrtf(var + EPS_C) * gamma[co] + beta[co];
    }
}

__global__ void k_copyH(const float* __restrict__ H, float* dst) {
    int t = threadIdx.x;
    if (t < Gn * 9) dst[t] = H[t];
}

// ---------------------------------------------------------------------------
extern "C" void kernel_launch(void* const* d_in, const int* in_sizes, int n_in,
                              void* d_out, int out_size, void* d_ws, size_t ws_size,
                              hipStream_t stream) {
    const float* x     = (const float*)d_in[0];
    const float* H     = (const float*)d_in[1];
    const float* Hr1   = (const float*)d_in[2];
    const float* Hr2   = (const float*)d_in[3];
    const float* Hrr   = (const float*)d_in[4];
    const float* Wg1   = (const float*)d_in[5];
    const float* Ws1   = (const float*)d_in[6];
    const float* Wg2   = (const float*)d_in[7];
    const float* Ws2   = (const float*)d_in[8];
    const float* Wres  = (const float*)d_in[9];
    const float* gamma = (const float*)d_in[10];
    const float* beta  = (const float*)d_in[11];
    float* out = (float*)d_out;                  // [VOLY y][108 H]

    char* ws = (char*)d_ws;
    size_t off = 0;
    auto alloc = [&](size_t bytes) -> void* {
        void* p = ws + off;
        off = (off + bytes + 255) & ~(size_t)255;
        return p;
    };
    float* zbuf = (float*)alloc(VOLY * sizeof(float));      // 100 MB scratch
    bf16*  xbT  = (bf16*) alloc(VOLX * sizeof(bf16));       // x^T  [b][m][k]
    bf16*  t1bT = (bf16*) alloc(VOLY * sizeof(bf16));       // t1^T [b][m][n]
    bf16*  A1   = (bf16*) alloc((size_t)NG * K1n * sizeof(bf16));
    bf16*  A2   = (bf16*) alloc((size_t)NG * K2n * sizeof(bf16));
    bf16*  Ares = (bf16*) alloc((size_t)NG * K1n * sizeof(bf16));
    float* w1   = (float*)alloc(Gn * Gn * Rn * sizeof(float));
    float* w2   = (float*)alloc(Gn * Gn * Rn * sizeof(float));
    float* wr   = (float*)alloc(Gn * Gn * Rn * sizeof(float));
    float* Kr1  = (float*)alloc(Cout * Gn * 27 * sizeof(float));
    float* Kr2  = (float*)alloc(Cout * Gn * 27 * sizeof(float));
    float* stats = (float*)alloc(320 * sizeof(float));
    float* st1 = stats;
    float* st2 = stats + 128;
    float* bst = stats + 256;

    k_zero<<<1, 512, 0, stream>>>(stats, 320);
    k_prep_small<<<1, 256, 0, stream>>>(H, Hr1, Hr2, Hrr, Ws1, Ws2,
                                        w1, w2, wr, Kr1, Kr2);
    k_prep_A<<<1152, 256, 0, stream>>>(Wg1, Wg2, Wres, w1, w2, wr, A1, A2, Ares);
    // x -> bf16, transposed to [b][m][k]
    k_cvtT<<<dim3(S3 / 32, K1n / 32, Bsz), 256, 0, stream>>>(x, xbT);
    // z1 = groupmix1(x)   [TDM + WMMA]
    k_gemm<<<dim3(S3 / 32, NG / 128, Bsz), 256, 0, stream>>>(A1, xbT, zbuf, K1n);
    k_dwconv<<<dim3(S3 / 256, Bsz * NG), 256, 0, stream>>>(zbuf, Kr1, out);
    // instance norm 1 + ReLU -> transposed bf16 operand for GEMM2
    k_instats<<<dim3(24, Bsz * Cout), 256, 0, stream>>>(out, st1);
    k_in_transpose<<<dim3(S3 / 32, NG / 32, Bsz), 256, 0, stream>>>(out, st1, t1bT);
    // z2 = groupmix2(t1)  [TDM + WMMA]
    k_gemm<<<dim3(S3 / 32, NG / 128, Bsz), 256, 0, stream>>>(A2, t1bT, zbuf, K2n);
    k_dwconv<<<dim3(S3 / 256, Bsz * NG), 256, 0, stream>>>(zbuf, Kr2, out);
    // instance norm 2 + ReLU (fp32 in place)
    k_instats<<<dim3(24, Bsz * Cout), 256, 0, stream>>>(out, st2);
    k_inapply_f32<<<4096, 256, 0, stream>>>(out, st2);
    // res = groupmix_res(x)  [TDM + WMMA]
    k_gemm<<<dim3(S3 / 32, NG / 128, Bsz), 256, 0, stream>>>(Ares, xbT, zbuf, K1n);
    // y += res with batchnorm stats, then apply + H tail
    k_add_bnstats<<<dim3(96, Bsz * Cout), 256, 0, stream>>>(out, zbuf, bst);
    k_bnapply<<<4096, 256, 0, stream>>>(out, bst, gamma, beta);
    k_copyH<<<1, 128, 0, stream>>>(H, out + VOLY);
}
